// FBPinn_70179765616816
// MI455X (gfx1250) — matrix operand order repaired and checked by reading, not compiled, not run on hardware
//
#include <hip/hip_runtime.h>
#include <hip/hip_bf16.h>

// ---------------------------------------------------------------------------
// FBPinn forward: 32 windows x (1->128->128->128->1 tanh MLP) + sigmoid blend.
// ~105 GFLOP vs ~5 MB unique data -> compute-bound -> f16 WMMA, f32 accumulate.
// One block = 128 points = 4 wave32, each wave owns an M=32 row tile so every
// LDS B-fragment (1KB) feeds TWO v_wmma_f32_16x16x32_f16 ops (LDS read BW is
// the binding resource: ~64B/clk/SIMD vs ~128B/clk demanded by M=16 at XDL
// rate). 391 blocks, ~105KB LDS -> 3 blocks/WGP (12 waves) for any plausible
// WGP count. Weights pre-converted to f16 in d_ws and staged with
// global_load_async_to_lds_b128, double-buffered so the DMA overlaps the
// previous layer's WMMA. tanh -> v_tanh_f32. No atomics: deterministic.
// ---------------------------------------------------------------------------

typedef __attribute__((ext_vector_type(16))) _Float16 v16h;
typedef __attribute__((ext_vector_type(8)))  _Float16 v8h;
typedef __attribute__((ext_vector_type(4)))  _Float16 v4h;
typedef __attribute__((ext_vector_type(8)))  float    v8f;

#define FB_NW     32
#define FB_NEUR   128
#define FB_LDH    136      // padded LDS row stride in halfs (bank shift 4 dwords/row)
#define FB_PTS    128      // points per block (4 waves x M=32)
#define FB_THR    128      // threads per block
#define FB_ISIG   50.0f    // 1/sigma
#define FB_LOG2E  1.4426950408889634f

__device__ __forceinline__ float fb_tanh(float x) {
#if __has_builtin(__builtin_amdgcn_tanhf)
    return __builtin_amdgcn_tanhf(x);            // v_tanh_f32 (confirmed in asm)
#else
    float e = __builtin_amdgcn_exp2f(x * (2.0f * FB_LOG2E));
    return 1.0f - 2.0f * __builtin_amdgcn_rcpf(e + 1.0f);
#endif
}
__device__ __forceinline__ float fb_sigmoid(float z) {
    return __builtin_amdgcn_rcpf(1.0f + __builtin_amdgcn_exp2f(-z * FB_LOG2E));
}

// ---- async global->LDS copy (gfx1250; ASYNCcnt) ----
__device__ __forceinline__ void fb_async_b128(unsigned ldsOff, const void* g) {
    asm volatile("global_load_async_to_lds_b128 %0, %1, off"
                 :: "v"(ldsOff), "v"(g) : "memory");
}
__device__ __forceinline__ void fb_wait_async0() {
    asm volatile("s_wait_asynccnt 0x0" ::: "memory");
}

// Async-copy one 128x128 f16 weight tile ([n][k] contiguous in global) into
// padded LDS rows [n][FB_LDH]. 2048 16B chunks / 128 threads = 16 per thread.
__device__ __forceinline__ void fb_issue_wt(const _Float16* __restrict__ src,
                                            _Float16* dstLds, int tid) {
    unsigned base = (unsigned)(uintptr_t)dstLds;   // generic->LDS offset (addr[31:0])
    #pragma unroll
    for (int i = 0; i < 16; ++i) {
        int c  = tid + i * FB_THR;
        int n  = c >> 4;
        int k8 = (c & 15) << 3;
        fb_async_b128(base + (unsigned)(n * FB_LDH + k8) * 2u, src + n * 128 + k8);
    }
}

// Fallback: synchronous f32->f16 transpose staging (lane-strided global reads, b64 stores).
__device__ __forceinline__ void fb_stage_f32(const float* __restrict__ Wg,
                                             _Float16* WT, int tid) {
    #pragma unroll 4
    for (int i = 0; i < 32; ++i) {
        int c  = tid + i * FB_THR;
        int n  = c >> 5;
        int k4 = (c & 31) << 2;
        v4h h4;
        #pragma unroll
        for (int j = 0; j < 4; ++j) h4[j] = (_Float16)Wg[(k4 + j) * 128 + n];
        *(v4h*)(WT + n * FB_LDH + k4) = h4;
    }
}

// ---- prologue: W_hid f32 [l][w][k][n] -> f16 transposed cache [l][w][n][k] ----
__global__ __launch_bounds__(256) void fb_convert_weights(
    const float* __restrict__ W_hid, _Float16* __restrict__ WT16, int total) {
    int o = blockIdx.x * 256 + threadIdx.x;
    if (o >= total) return;
    int k  = o & 127;
    int n  = (o >> 7) & 127;
    int lw = o >> 14;
    WT16[o] = (_Float16)W_hid[(size_t)lw * 16384 + k * 128 + n];
}

__global__ __launch_bounds__(FB_THR) void fbpinn_wmma_kernel(
    const float* __restrict__ x,
    const float* __restrict__ W_in,  const float* __restrict__ b_in,
    const float* __restrict__ W_hid, const float* __restrict__ b_hid,
    const float* __restrict__ W_out, const float* __restrict__ b_out,
    const _Float16* __restrict__ WT16,   // f16 weight cache in d_ws (or nullptr)
    float* __restrict__ out, int npts)
{
    extern __shared__ char smem_raw[];
    _Float16* h    = (_Float16*)smem_raw;               // [128][FB_LDH] activations
    _Float16* WTb0 = h    + FB_PTS  * FB_LDH;           // weight buffer 0
    _Float16* WTb1 = WTb0 + FB_NEUR * FB_LDH;           // weight buffer 1
    float*    xs   = (float*)(WTb1 + FB_NEUR * FB_LDH); // [128]

    const int tid    = threadIdx.x;
    const int lane   = tid & 31;
    const int wave   = tid >> 5;            // 0..3
    const int ptbase = wave * 32;           // 32 rows (points) owned by this wave
    const int mrow   = lane & 15;
    const int hi     = (lane >> 4) & 1;
    const int gbase  = blockIdx.x * FB_PTS;
    const bool use_cache = (WT16 != nullptr);           // uniform

    {
        int g = gbase + tid;
        xs[tid] = (g < npts) ? x[g] : 0.0f;             // pad ragged tile with x=0
    }
    float accp = 0.0f;                                  // point tid's accumulator

    // Prefetch window 0 / layer 0 weights while doing the input layer.
    if (use_cache) fb_issue_wt(WT16, WTb0, tid);
    __syncthreads();

    const float width = 1.0f / (float)FB_NW;

    for (int w = 0; w < FB_NW; ++w) {
        float l    = (w == 0)         ? 0.0f : ((float)w - 0.125f) * width;
        float r    = (w == FB_NW - 1) ? 1.0f : ((float)w + 1.125f) * width;
        float mean = 0.5f * (l + r);
        float istd = 2.0f / (r - l);

        // -------- input layer: h[p][n] = tanh(xn*W_in + b_in) --------
        for (int idx = tid; idx < FB_PTS * FB_NEUR; idx += FB_THR) {
            int p = idx >> 7, n = idx & 127;
            float xn = (xs[p] - mean) * istd;
            h[p * FB_LDH + n] =
                (_Float16)fb_tanh(xn * W_in[w * FB_NEUR + n] + b_in[w * FB_NEUR + n]);
        }

        // -------- two 128x128 hidden layers on WMMA, double-buffered weights ----
        for (int layer = 0; layer < 2; ++layer) {
            _Float16* WT = (layer == 0) ? WTb0 : WTb1;
            const float* bg = b_hid + (layer * FB_NW + w) * FB_NEUR;

            if (use_cache) {
                fb_wait_async0();      // this buffer's DMA (issued one stage ago) done
                __syncthreads();       // ...across all waves; also orders h (layer 0)
                if (layer == 0)        // overlap next stage's DMA with this GEMM
                    fb_issue_wt(WT16 + (size_t)(FB_NW + w) * 16384, WTb1, tid);
                else if (w + 1 < FB_NW)
                    fb_issue_wt(WT16 + (size_t)(w + 1) * 16384, WTb0, tid);
            } else {
                __syncthreads();
                fb_stage_f32(W_hid + (size_t)(layer * FB_NW + w) * 16384, WT, tid);
                __syncthreads();
            }

            // Preload all 8 A fragments (2 row tiles x full K=128) up front
            // -> in-place h update is hazard-free.
            // A layout (16x32 f16): a[i] <-> K = 32*ks + 16*(i/8) + 8*hi + (i%8)
            v16h afr[2][4];
            #pragma unroll
            for (int rt = 0; rt < 2; ++rt) {
                const _Float16* hrow = h + (ptbase + rt * 16 + mrow) * FB_LDH;
                #pragma unroll
                for (int ks = 0; ks < 4; ++ks) {
                    int K0 = ks * 32 + hi * 8;
                    v8h lo = *(const v8h*)(hrow + K0);
                    v8h up = *(const v8h*)(hrow + K0 + 16);
                    v16h a;
                    #pragma unroll
                    for (int i = 0; i < 8; ++i) { a[i] = lo[i]; a[i + 8] = up[i]; }
                    afr[rt][ks] = a;
                }
            }

            #pragma unroll
            for (int t = 0; t < 8; ++t) {
                int n = t * 16 + mrow;
                float bias = bg[n];
                const _Float16* wrow = WT + n * FB_LDH;
                v8f c0 = {};
                v8f c1 = {};
                #pragma unroll
                for (int ks = 0; ks < 4; ++ks) {
                    // B layout (32x16 f16): b[i] <-> K = 32*ks + 16*(i/8) + 8*hi + (i%8), col = lane&15
                    int K0 = ks * 32 + hi * 8;
                    v8h b0 = *(const v8h*)(wrow + K0);
                    v8h b1 = *(const v8h*)(wrow + K0 + 16);
                    v16h b;
                    #pragma unroll
                    for (int i = 0; i < 8; ++i) { b[i] = b0[i]; b[i + 8] = b1[i]; }
                    // one B fragment feeds two WMMAs (A-tile reuse)
                    c0 = __builtin_amdgcn_wmma_f32_16x16x32_f16(
                            false, afr[0][ks], false, b, (short)0, c0, false, false);
                    c1 = __builtin_amdgcn_wmma_f32_16x16x32_f16(
                            false, afr[1][ks], false, b, (short)0, c1, false, false);
                }
                #pragma unroll
                for (int rr = 0; rr < 8; ++rr) {
                    int row0 = ptbase + rr + hi * 8;
                    h[row0 * FB_LDH + n]        = (_Float16)fb_tanh(c0[rr] + bias);
                    h[(row0 + 16) * FB_LDH + n] = (_Float16)fb_tanh(c1[rr] + bias);
                }
            }
        }
        __syncthreads();   // h complete before output reads

        // -------- output layer (128->1) + sigmoid window blend (all threads) ----
        {
            const float* Wo = W_out + w * FB_NEUR;
            const _Float16* hp = h + tid * FB_LDH;
            float s = 0.0f;
            #pragma unroll 8
            for (int n = 0; n < FB_NEUR; ++n) s += (float)hp[n] * Wo[n];
            s += b_out[w];                                 // U_SD=1, U_MEAN=0
            float xv  = xs[tid];
            float win = fb_sigmoid((xv - (float)w * width) * FB_ISIG)
                      * fb_sigmoid(-(xv - (float)(w + 1) * width) * FB_ISIG);
            accp += win * s;
        }
        __syncthreads();   // h consumed before next window's input layer
    }

    {
        int g = gbase + tid;
        if (g < npts) out[g] = accp;
    }
}

extern "C" void kernel_launch(void* const* d_in, const int* in_sizes, int n_in,
                              void* d_out, int out_size, void* d_ws, size_t ws_size,
                              hipStream_t stream) {
    const float* x     = (const float*)d_in[0];
    const float* W_in  = (const float*)d_in[1];
    const float* b_in  = (const float*)d_in[2];
    const float* W_hid = (const float*)d_in[3];
    const float* b_hid = (const float*)d_in[4];
    const float* W_out = (const float*)d_in[5];
    const float* b_out = (const float*)d_in[6];
    float* out = (float*)d_out;

    int npts = in_sizes[0];
    int nblk = (npts + FB_PTS - 1) / FB_PTS;

    const int    wtotal  = 2 * FB_NW * FB_NEUR * FB_NEUR;        // 1,048,576 halfs
    const size_t wtbytes = (size_t)wtotal * 2;                   // 2 MB
    _Float16* WT16 = (ws_size >= wtbytes) ? (_Float16*)d_ws : nullptr;

    if (WT16) {
        fb_convert_weights<<<(wtotal + 255) / 256, 256, 0, stream>>>(W_hid, WT16, wtotal);
    }

    size_t shmem = (size_t)(FB_PTS + 2 * FB_NEUR) * FB_LDH * 2   // h + 2 weight buffers
                 + (size_t)FB_PTS * sizeof(float);               // xs
    fbpinn_wmma_kernel<<<nblk, FB_THR, shmem, stream>>>(
        x, W_in, b_in, W_hid, b_hid, W_out, b_out, WT16, out, npts);
}
